// Attention_gvtn_22041772163233
// MI455X (gfx1250) — matrix operand order repaired and checked by reading, not compiled, not run on hardware
//
#include <hip/hip_runtime.h>
#include <hip/hip_bf16.h>

// ---- problem constants (from reference) ----
#define BATCH 2
#define CH    64
#define NH    8
#define DH    8          // head dim = 64/8
#define HH    48
#define WW    48
#define L     (HH*WW)    // 2304
#define QSCALE 0.35355339059327373f  // 8^-0.5

typedef __attribute__((ext_vector_type(16))) _Float16 v16h;
typedef __attribute__((ext_vector_type(8)))  _Float16 v8h;
typedef __attribute__((ext_vector_type(8)))  float    v8f;
typedef __attribute__((ext_vector_type(4)))  unsigned int u32x4;
typedef __attribute__((ext_vector_type(8)))  int i32x8;
typedef __attribute__((ext_vector_type(4)))  int i32x4;

union VB { v16h v; v8h h8[2]; };
union H32 { _Float16 h[32]; float4 f4[4]; };

// ---------------------------------------------------------------------
// TDM: DMA a 64x64 f32 weight matrix (16KB) from global into LDS.
// D# built per CDNA5 ISA 8.3/8.4: 1D tile, data_size=4B,
// tensor_dim0 = tile_dim0 = 4096 elements, tensor_dim1 = tile_dim1 = 1.
// Issued by one wave; completion via S_WAIT_TENSORCNT.
// ---------------------------------------------------------------------
__device__ __forceinline__ void tdm_load_w16k(const float* gsrc, float* ldst) {
  const unsigned lo  = (unsigned)(uintptr_t)gsrc;
  const unsigned hi  = (unsigned)(((uintptr_t)gsrc) >> 32);
  const unsigned lds = (unsigned)(uintptr_t)ldst;   // LDS offset = addr[31:0]

  u32x4 g0;
  g0[0] = 1u;                                   // count=1, user descriptor
  g0[1] = lds;                                  // lds_addr[31:0]
  g0[2] = lo;                                   // global_addr[31:0]
  g0[3] = (hi & 0x01FFFFFFu) | (2u << 30);      // global_addr[56:32] | type=2

  i32x8 g1;
  g1[0] = (int)(2u << 16);                      // workgroup_mask=0, data_size=4B
  g1[1] = (int)(4096u << 16);                   // tensor_dim0[15:0] in [31:16]
  g1[2] = (int)(1u << 16);                      // tensor_dim0 hi=0, tensor_dim1=1
  g1[3] = (int)(4096u << 16);                   // tensor_dim1 hi=0, tile_dim0=4096
  g1[4] = 1;                                    // tile_dim1=1, tile_dim2=0
  g1[5] = 4096;                                 // tensor_dim0_stride low32
  g1[6] = 0;                                    // stride0 hi, stride1 lo
  g1[7] = 0;

  i32x4 z4 = {0, 0, 0, 0};
#if __clang_major__ >= 23
  i32x8 z8 = {0, 0, 0, 0, 0, 0, 0, 0};
  __builtin_amdgcn_tensor_load_to_lds(g0, g1, z4, z4, z8, 0);
#else
  __builtin_amdgcn_tensor_load_to_lds(g0, g1, z4, z4, 0);
#endif
}

// =====================================================================
// Kernel A: fused QKV 1x1 conv. Weights staged via TDM. Writes:
//   - scaled q (f32) to qout  (second half of d_out, the "q" return value)
//   - qpad/kpad: [b*NH+h][L][32] f16, entries 0..7 = head dims, 8..31 = 0
//   - vcl: [b*NH+h][16][L] f16, rows 0..7 = v head dims, 8..15 = 0
// =====================================================================
__global__ __launch_bounds__(256) void qkv_proj_kernel(
    const float* __restrict__ x,
    const float* __restrict__ Wq, const float* __restrict__ bq,
    const float* __restrict__ Wk, const float* __restrict__ bk,
    const float* __restrict__ Wv, const float* __restrict__ bv,
    float* __restrict__ qout,
    _Float16* __restrict__ qpad, _Float16* __restrict__ kpad,
    _Float16* __restrict__ vcl)
{
  __shared__ float sW[3][CH*CH];
  __shared__ float sb[3][CH];
  if (threadIdx.x < 32) {                       // wave 0 drives the TDM
    tdm_load_w16k(Wq, &sW[0][0]);
    tdm_load_w16k(Wk, &sW[1][0]);
    tdm_load_w16k(Wv, &sW[2][0]);
    __builtin_amdgcn_s_wait_tensorcnt(0);
  }
  if (threadIdx.x < CH) {
    sb[0][threadIdx.x] = bq[threadIdx.x];
    sb[1][threadIdx.x] = bk[threadIdx.x];
    sb[2][threadIdx.x] = bv[threadIdx.x];
  }
  __syncthreads();

  const int idx = blockIdx.x * 256 + threadIdx.x;   // [0, B*L)
  const int b = idx / L;
  const int p = idx - b * L;

  float xv[CH];
#pragma unroll
  for (int c = 0; c < CH; ++c) xv[c] = x[((size_t)b * CH + c) * L + p];

  for (int h = 0; h < NH; ++h) {
    H32 qh, kh;
    float vvals[DH];
#pragma unroll
    for (int d = 0; d < DH; ++d) {
      const int o = h * DH + d;
      float aq = sb[0][o], ak = sb[1][o], av = sb[2][o];
      const float* wq = &sW[0][o * CH];
      const float* wk = &sW[1][o * CH];
      const float* wv = &sW[2][o * CH];
#pragma unroll
      for (int c = 0; c < CH; ++c) {
        const float xc = xv[c];
        aq = fmaf(wq[c], xc, aq);
        ak = fmaf(wk[c], xc, ak);
        av = fmaf(wv[c], xc, av);
      }
      aq *= QSCALE;                                  // reference scales q in place
      qout[((size_t)b * CH + o) * L + p] = aq;
      qh.h[d] = (_Float16)aq;
      kh.h[d] = (_Float16)ak;
      vvals[d] = av;
    }
#pragma unroll
    for (int j = DH; j < 32; ++j) { qh.h[j] = (_Float16)0.f; kh.h[j] = (_Float16)0.f; }

    const size_t r = ((size_t)(b * NH + h) * L + p) * 32;
    float4* qd = (float4*)(qpad + r);
    float4* kd = (float4*)(kpad + r);
#pragma unroll
    for (int j = 0; j < 4; ++j) { qd[j] = qh.f4[j]; kd[j] = kh.f4[j]; }

#pragma unroll
    for (int c = 0; c < 16; ++c) {
      vcl[((size_t)(b * NH + h) * 16 + c) * L + p] =
          (c < DH) ? (_Float16)vvals[c] : (_Float16)0.f;
    }
  }
}

// =====================================================================
// Kernel B: per-column (k) softmax stats over the QUERY axis.
// One wave owns 16 k-columns of one (b,h); loops over 144 q-tiles with
// v_wmma_f32_16x16x32_f16 and maintains online max/sum per column.
// D-layout: lane l holds (q = g + 8*(l>=16), k = l%16) for VGPR g.
// =====================================================================
__global__ __launch_bounds__(256) void col_softmax_stats_kernel(
    const _Float16* __restrict__ qpad, const _Float16* __restrict__ kpad,
    float* __restrict__ mbuf, float* __restrict__ rzbuf)
{
  const int lane = threadIdx.x & 31;
  const int w    = threadIdx.x >> 5;
  const int tile = blockIdx.x * 8 + w;              // [0, 16*144)
  const int bh   = tile / (L / 16);
  const int k0   = (tile % (L / 16)) * 16;
  const int half = lane >> 4;
  const int col  = lane & 15;

  // B operand: K-tile [32 x 16], element (r,n) = kpad[(k0+n)*32 + r]
  const _Float16* krow = kpad + ((size_t)bh * L + (k0 + col)) * 32 + 16 * half;
  VB kb;
  kb.h8[0] = *(const v8h*)(krow);
  kb.h8[1] = *(const v8h*)(krow + 8);

  float m = -3.0e38f, Z = 0.f;
  const _Float16* qbase = qpad + (size_t)bh * L * 32;

  for (int q0 = 0; q0 < L; q0 += 16) {
    // A operand: Q^T tile [16 x 32], lane row = lane%16
    const _Float16* qrow = qbase + ((size_t)(q0 + col)) * 32;
    VB qa;
    qa.h8[0] = *(const v8h*)(qrow + 16 * half);
    qa.h8[1] = *(const v8h*)(qrow + 16 * half + 8);

    v8f s = {};
    s = __builtin_amdgcn_wmma_f32_16x16x32_f16(false, qa.v, false, kb.v,
                                               (short)0, s, false, false);
    float tm = s[0];
#pragma unroll
    for (int g = 1; g < 8; ++g) tm = fmaxf(tm, s[g]);
    const float nm = fmaxf(m, tm);
    float zs = 0.f;
#pragma unroll
    for (int g = 0; g < 8; ++g) zs += __expf(s[g] - nm);
    Z = Z * __expf(m - nm) + zs;
    m = nm;
  }

  // combine the two lane-halves (lane l and l+16 hold the same k-column)
  __shared__ float red[8][2][32];
  red[w][0][lane] = m;
  red[w][1][lane] = Z;
  asm volatile("s_wait_dscnt 0" ::: "memory");
  if (lane < 16) {
    const float m2 = red[w][0][lane + 16];
    const float Z2 = red[w][1][lane + 16];
    const float M  = fmaxf(m, m2);
    const float Zt = Z * __expf(m - M) + Z2 * __expf(m2 - M);
    mbuf [(size_t)bh * L + k0 + lane] = M;
    rzbuf[(size_t)bh * L + k0 + lane] = 1.f / Zt;
  }
}

// =====================================================================
// Kernel C: o^T[q,c] = sum_k P[q,k] * v[c,k].  One wave owns a 16-q tile
// of one (b,h).  Per 32-k chunk: 2 WMMAs recompute S, P built in f16 via
// a wave-private LDS tile to convert D-layout -> A-layout, 1 WMMA
// accumulates O.  f32 accumulation throughout.
// =====================================================================
__global__ __launch_bounds__(256) void attn_apply_kernel(
    const _Float16* __restrict__ qpad, const _Float16* __restrict__ kpad,
    const _Float16* __restrict__ vcl,
    const float* __restrict__ mbuf, const float* __restrict__ rzbuf,
    float* __restrict__ obuf)
{
  const int lane = threadIdx.x & 31;
  const int w    = threadIdx.x >> 5;
  const int tile = blockIdx.x * 8 + w;
  const int bh   = tile / (L / 16);
  const int q0   = (tile % (L / 16)) * 16;
  const int half = lane >> 4;
  const int col  = lane & 15;
  const int b    = bh >> 3;
  const int h    = bh & 7;

  __shared__ alignas(16) _Float16 pbuf[8][16][32];  // per-wave 1KB P tile

  // fixed A operand: Q^T tile for these 16 queries
  const _Float16* qrow = qpad + ((size_t)bh * L + (q0 + col)) * 32;
  VB qa;
  qa.h8[0] = *(const v8h*)(qrow + 16 * half);
  qa.h8[1] = *(const v8h*)(qrow + 16 * half + 8);

  const float* mcol = mbuf  + (size_t)bh * L;
  const float* rcol = rzbuf + (size_t)bh * L;

  v8f acc = {};
  for (int k0 = 0; k0 < L; k0 += 32) {
    if (k0 + 32 < L) {
      __builtin_prefetch(kpad + ((size_t)bh * L + (k0 + 32 + col)) * 32, 0, 1);
      __builtin_prefetch(vcl + ((size_t)bh * 16 + col) * L + k0 + 32, 0, 1);
    }
#pragma unroll
    for (int t = 0; t < 2; ++t) {
      const int kc = k0 + 16 * t + col;
      const _Float16* krow = kpad + ((size_t)bh * L + kc) * 32 + 16 * half;
      VB kb;
      kb.h8[0] = *(const v8h*)(krow);
      kb.h8[1] = *(const v8h*)(krow + 8);

      v8f s = {};
      s = __builtin_amdgcn_wmma_f32_16x16x32_f16(false, qa.v, false, kb.v,
                                                 (short)0, s, false, false);
      const float mk = mcol[kc];
      const float rz = rcol[kc];
#pragma unroll
      for (int g = 0; g < 8; ++g) {
        const float pv = __expf(s[g] - mk) * rz;
        pbuf[w][g + 8 * half][16 * t + col] = (_Float16)pv;
      }
    }
    asm volatile("s_wait_dscnt 0" ::: "memory");

    // A operand: P tile [16 x 32] from LDS (A layout)
    VB pa;
    pa.h8[0] = *(const v8h*)&pbuf[w][col][16 * half];
    pa.h8[1] = *(const v8h*)&pbuf[w][col][16 * half + 8];

    // B operand: V^T chunk [32 x 16], element (r,n) = vcl[n][k0+r]
    const _Float16* vrow = vcl + ((size_t)bh * 16 + col) * L + k0 + 16 * half;
    VB vb;
    vb.h8[0] = *(const v8h*)(vrow);
    vb.h8[1] = *(const v8h*)(vrow + 8);

    acc = __builtin_amdgcn_wmma_f32_16x16x32_f16(false, pa.v, false, vb.v,
                                                 (short)0, acc, false, false);
    asm volatile("" ::: "memory");  // keep LDS WAR ordering across iterations
  }

  // D layout: lane holds (q = g + 8*half, c = col); c < 8 valid
  if (col < DH) {
    float* dst = obuf + ((size_t)(b * CH) + h * DH + col) * L + q0;
#pragma unroll
    for (int g = 0; g < 8; ++g) dst[g + 8 * half] = acc[g];
  }
}

// =====================================================================
// Kernel D: output 1x1 conv:  out[b,o,p] = sum_c Wo[o,c]*obuf[b,c,p]+bo
// Wo staged via TDM.
// =====================================================================
__global__ __launch_bounds__(256) void out_proj_kernel(
    const float* __restrict__ obuf,
    const float* __restrict__ Wo, const float* __restrict__ bo,
    float* __restrict__ out)
{
  __shared__ float sW[CH*CH];
  __shared__ float sb[CH];
  if (threadIdx.x < 32) {
    tdm_load_w16k(Wo, &sW[0]);
    __builtin_amdgcn_s_wait_tensorcnt(0);
  }
  if (threadIdx.x < CH) sb[threadIdx.x] = bo[threadIdx.x];
  __syncthreads();

  const int idx = blockIdx.x * 256 + threadIdx.x;
  const int b = idx / L;
  const int p = idx - b * L;

  float xv[CH];
#pragma unroll
  for (int c = 0; c < CH; ++c) xv[c] = obuf[((size_t)b * CH + c) * L + p];

  for (int o = 0; o < CH; ++o) {
    float a = sb[o];
    const float* wrow = &sW[o * CH];
#pragma unroll
    for (int c = 0; c < CH; ++c) a = fmaf(wrow[c], xv[c], a);
    out[((size_t)b * CH + o) * L + p] = a;
  }
}

// =====================================================================
extern "C" void kernel_launch(void* const* d_in, const int* in_sizes, int n_in,
                              void* d_out, int out_size, void* d_ws, size_t ws_size,
                              hipStream_t stream) {
  const float* x  = (const float*)d_in[0];
  const float* Wq = (const float*)d_in[1];
  const float* bq = (const float*)d_in[2];
  const float* Wk = (const float*)d_in[3];
  const float* bk = (const float*)d_in[4];
  const float* Wv = (const float*)d_in[5];
  const float* bv = (const float*)d_in[6];
  const float* Wo = (const float*)d_in[7];
  const float* bo = (const float*)d_in[8];

  float* out  = (float*)d_out;                       // [B,64,48,48]
  float* qout = (float*)d_out + (size_t)BATCH*CH*L;  // [B,64,48,48] (scaled q)

  // workspace layout (bytes)
  char* ws = (char*)d_ws;
  const size_t QPAD_B = (size_t)BATCH*NH*L*32*sizeof(_Float16);  // 2,359,296
  const size_t VCL_B  = (size_t)BATCH*NH*16*L*sizeof(_Float16);  // 1,179,648
  const size_t MZ_B   = (size_t)BATCH*NH*L*sizeof(float);        //   147,456
  _Float16* qpad = (_Float16*)(ws);
  _Float16* kpad = (_Float16*)(ws + QPAD_B);
  _Float16* vcl  = (_Float16*)(ws + 2*QPAD_B);
  float*    mbuf = (float*)   (ws + 2*QPAD_B + VCL_B);
  float*    rzb  = (float*)   (ws + 2*QPAD_B + VCL_B + MZ_B);
  float*    obuf = (float*)   (ws + 2*QPAD_B + VCL_B + 2*MZ_B);

  const int posBlocks  = (BATCH * L) / 256;              // 18
  const int tileBlocks = (BATCH * NH * (L / 16)) / 8;    // 288

  qkv_proj_kernel<<<posBlocks, 256, 0, stream>>>(
      x, Wq, bq, Wk, bk, Wv, bv, qout, qpad, kpad, vcl);

  col_softmax_stats_kernel<<<tileBlocks, 256, 0, stream>>>(
      qpad, kpad, mbuf, rzb);

  attn_apply_kernel<<<tileBlocks, 256, 0, stream>>>(
      qpad, kpad, vcl, mbuf, rzb, obuf);

  out_proj_kernel<<<posBlocks, 256, 0, stream>>>(
      obuf, Wo, bo, out);
}
